// Hybrid_84499186581845
// MI455X (gfx1250) — matrix operand order repaired and checked by reading, not compiled, not run on hardware
//
#include <hip/hip_runtime.h>
#include <hip/hip_bf16.h>

// ---------------------------------------------------------------------------
// MI455X (gfx1250) hybrid CNN-encoder / graph-decoder forward pass.
// GEMM-shaped work (implicit-GEMM convs, cheb layers) on
// v_wmma_f32_16x16x32_f16 with LDS-staged f16 tiles; BN folded to per-channel
// scale/shift; batched (two-phase, unrolled) staging loads; register
// double-buffered K pipeline in the GEMM fast path; all buffers L2-resident.
// ---------------------------------------------------------------------------

typedef __attribute__((ext_vector_type(16))) _Float16 v16h;
typedef __attribute__((ext_vector_type(2)))  _Float16 h2;
typedef __attribute__((ext_vector_type(8)))  float    v8f;

#define EPSV 1e-5f
#define NWAVES 4   // waves per block for WMMA kernels (128 threads)

static inline int ceilDiv(int a, int b) { return (a + b - 1) / b; }

// K index (0..31) of element e of the 16-bit A fragment (16x32), lane-half kh.
__device__ __forceinline__ int a_k_of(int e, int kh) {
  int vg = e >> 1;
  int base = (vg < 4) ? (2 * vg) : (16 + 2 * (vg - 4));
  return base + 8 * kh + (e & 1);
}

// ---------------------------------------------------------------------------
// Per-channel batch stats folded with affine params:
// scale = g * rsqrt(var+eps), shift = b - mean*scale.
// ---------------------------------------------------------------------------
__global__ void bn_stats_fold_kernel(const float* __restrict__ in,
                                     const float* __restrict__ g, const float* __restrict__ bb,
                                     float* __restrict__ scale, float* __restrict__ shift,
                                     int Bn, int C, int HW2) {
  __shared__ float s1[256];
  __shared__ float s2[256];
  int c = blockIdx.x;
  int tid = threadIdx.x;
  long long Ntot = (long long)Bn * HW2;
  float a = 0.f, b = 0.f;
  for (long long t = tid; t < Ntot; t += 256) {
    int bidx = (int)(t / HW2);
    int i    = (int)(t - (long long)bidx * HW2);
    float v = in[((long long)bidx * C + c) * HW2 + i];
    a += v; b += v * v;
  }
  s1[tid] = a; s2[tid] = b;
  __syncthreads();
  for (int s = 128; s > 0; s >>= 1) {
    if (tid < s) { s1[tid] += s1[tid + s]; s2[tid] += s2[tid + s]; }
    __syncthreads();
  }
  if (tid == 0) {
    float m = s1[0] / (float)Ntot;
    float v = s2[0] / (float)Ntot - m * m;
    float inv = rsqrtf(v + EPSV);
    float sc = g[c] * inv;
    scale[c] = sc;
    shift[c] = bb[c] - m * sc;
  }
}

// ---------------------------------------------------------------------------
// Implicit-GEMM conv, templated on kernel size (constant-divisor k decompose).
// Batched two-phase gathers; weight tile staged in LDS as f16; fused
// relu(scale*x+shift); epilogue transposed via LDS for x-contiguous stores.
// ---------------------------------------------------------------------------
template <int KS, int PAD>
__global__ void conv_wmma_kernel(const float* __restrict__ in, const float* __restrict__ w,
                                 const float* __restrict__ bias, float* __restrict__ out,
                                 const float* __restrict__ bnS, const float* __restrict__ bnT,
                                 int Bn, int Cin, int H, int Wd, int Cout, int preact) {
  __shared__ _Float16 Bs[16 * 40];            // 16 N x 32 K, padded stride 40
  __shared__ float    CT[NWAVES][16 * 17];    // per-wave transpose tile
  const int tid  = threadIdx.x;
  const int lane = tid & 31;
  const int wave = tid >> 5;
  const int kh   = lane >> 4;
  const int l16  = lane & 15;
  const int brr  = tid >> 5;                  // staging row base (+4 per step)
  const int bkk  = tid & 31;                  // staging k
  const int HW2  = H * Wd;
  const int M    = Bn * HW2;
  const int mtile = (blockIdx.x * NWAVES + wave) * 16;
  const int ntile = blockIdx.y * 16;
  const int Ktot  = Cin * KS * KS;

  // per-lane output coordinate (A row) -- loop invariant
  const int m  = mtile + l16;
  const int mc = (m < M) ? m : 0;
  const int b   = mc / HW2;
  const int rem = mc - b * HW2;
  const int y = rem / Wd;
  const int x = rem - y * Wd;
  const float* inb = in + (long long)b * Cin * HW2;

  v8f acc = {};
  for (int kb = 0; kb < Ktot; kb += 32) {
    // ---- weight tile staging: batched loads, then LDS stores ----
    float wv[4];
    int   wok[4];
#pragma unroll
    for (int i = 0; i < 4; ++i) {
      int gk = kb + bkk;
      int gn = ntile + brr + 4 * i;
      wok[i] = (gk < Ktot) & (gn < Cout);
      long long o = wok[i] ? ((long long)gn * Ktot + gk) : 0;
      wv[i] = w[o];
    }
#pragma unroll
    for (int i = 0; i < 4; ++i)
      Bs[(brr + 4 * i) * 40 + bkk] = (_Float16)(wok[i] ? wv[i] : 0.f);
    __syncthreads();

    // ---- A fragment: predicates/addresses first, batched loads, then BN ----
    float av[16];
    int   aok[16];
    int   acin[16];
#pragma unroll
    for (int e = 0; e < 16; ++e) {
      int k = kb + a_k_of(e, kh);
      int ok, cin;
      long long offl;
      if (KS == 1) {
        cin = (k < Ktot) ? k : (Ktot - 1);
        ok = (k < Ktot) & (m < M);
        offl = ok ? ((long long)cin * HW2 + y * Wd + x) : 0;
      } else {
        int c0 = k / (KS * KS);             // constant divisor
        int r  = k - c0 * (KS * KS);
        int iy = y + r / KS - PAD;
        int ix = x + r % KS - PAD;
        ok = (k < Ktot) & (m < M) & (iy >= 0) & (iy < H) & (ix >= 0) & (ix < Wd);
        cin = (c0 < Cin) ? c0 : (Cin - 1);
        offl = ok ? ((long long)cin * HW2 + iy * Wd + ix) : 0;
      }
      aok[e] = ok;
      acin[e] = cin;
      av[e] = inb[offl];                    // unconditional, clamped address
    }
    v16h afrag, bfrag;
#pragma unroll
    for (int e = 0; e < 16; ++e) {
      float va = av[e];
      if (preact) {
        va = va * bnS[acin[e]] + bnT[acin[e]];
        va = va > 0.f ? va : 0.f;
      }
      afrag[e] = (_Float16)(aok[e] ? va : 0.f);
      bfrag[e] = Bs[l16 * 40 + a_k_of(e, kh)];
    }
    acc = __builtin_amdgcn_wmma_f32_16x16x32_f16(false, afrag, false, bfrag,
                                                 (short)0, acc, false, false);
    __syncthreads();
  }
  // transpose so stores walk contiguous x per output channel
  float* ct = CT[wave];
#pragma unroll
  for (int r = 0; r < 8; ++r) ct[(kh * 8 + r) * 17 + l16] = acc[r];
  __syncthreads();
#pragma unroll
  for (int r = 0; r < 8; ++r) {
    int n = ntile + kh * 8 + r;
    if (m < M && n < Cout) {
      float v = ct[l16 * 17 + kh * 8 + r] + (bias ? bias[n] : 0.f);
      out[((long long)(b * Cout + n) * H + y) * Wd + x] = v;
    }
  }
}

// ---------------------------------------------------------------------------
// WMMA GEMM: out[M,ldo] = A[M,lda] * Wt[N,K]^T + bias.
// Fast path (lda even, K%32==0, full M tile): float2 staging with register
// double-buffering -- next chunk's global loads issued while current chunk's
// WMMA runs from LDS. Generic path: batched select-based staging.
// ---------------------------------------------------------------------------
__global__ void gemm_wmma_kernel(const float* __restrict__ A, int lda,
                                 const float* __restrict__ Wt,
                                 const float* __restrict__ bias,
                                 float* __restrict__ out, int ldo,
                                 int M, int N, int K) {
  __shared__ _Float16 As[64 * 40];
  __shared__ _Float16 Bs[16 * 40];
  const int tid  = threadIdx.x;
  const int lane = tid & 31;
  const int wave = tid >> 5;
  const int kh   = lane >> 4;
  const int l16  = lane & 15;
  const int mtile0 = blockIdx.x * (NWAVES * 16);
  const int mtile  = mtile0 + wave * 16;
  const int ntile  = blockIdx.y * 16;

  v8f acc = {};
  const bool allFast = ((lda & 1) == 0) && ((K & 31) == 0) && (mtile0 + 64 <= M);

  if (allFast) {
    const int arr = tid >> 4;             // A/B row base (+8 per step)
    const int akp = (tid & 15) << 1;      // k pair offset
    const float* aBase = A  + (long long)(mtile0 + arr) * lda + akp;
    const float* bBase = Wt + (long long)(ntile  + arr) * K   + akp;
    const long long aStep = 8LL * lda;
    const long long bStep = 8LL * K;
    _Float16* asBase = &As[arr * 40 + akp];
    _Float16* bsBase = &Bs[arr * 40 + akp];

    float2 pa[8];
    float2 pb[2];
#pragma unroll
    for (int i = 0; i < 8; ++i) pa[i] = *(const float2*)(aBase + i * aStep);
#pragma unroll
    for (int i = 0; i < 2; ++i) pb[i] = *(const float2*)(bBase + i * bStep);

    for (int kb = 0; kb < K; kb += 32) {
#pragma unroll
      for (int i = 0; i < 8; ++i) {
        h2 hv = { (_Float16)pa[i].x, (_Float16)pa[i].y };
        *(h2*)(asBase + i * 320) = hv;    // 8 rows * 40
      }
#pragma unroll
      for (int i = 0; i < 2; ++i) {
        h2 hv = { (_Float16)pb[i].x, (_Float16)pb[i].y };
        *(h2*)(bsBase + i * 320) = hv;
      }
      __syncthreads();
      if (kb + 32 < K) {                  // prefetch next chunk into registers
#pragma unroll
        for (int i = 0; i < 8; ++i) pa[i] = *(const float2*)(aBase + (kb + 32) + i * aStep);
#pragma unroll
        for (int i = 0; i < 2; ++i) pb[i] = *(const float2*)(bBase + (kb + 32) + i * bStep);
      }
      v16h afrag, bfrag;
#pragma unroll
      for (int e = 0; e < 16; ++e) {
        afrag[e] = As[(wave * 16 + l16) * 40 + a_k_of(e, kh)];
        bfrag[e] = Bs[l16 * 40 + a_k_of(e, kh)];
      }
      acc = __builtin_amdgcn_wmma_f32_16x16x32_f16(false, afrag, false, bfrag,
                                                   (short)0, acc, false, false);
      __syncthreads();
    }
  } else {
    const int brr = tid >> 5;             // row base (+4 per step)
    const int bkk = tid & 31;
    for (int kb = 0; kb < K; kb += 32) {
      float va[16];
      int   okA[16];
#pragma unroll
      for (int i = 0; i < 16; ++i) {
        int gm = mtile0 + brr + 4 * i;
        int gk = kb + bkk;
        okA[i] = (gm < M) & (gk < K);
        long long o = okA[i] ? ((long long)gm * lda + gk) : 0;
        va[i] = A[o];
      }
      float vb[4];
      int   okB[4];
#pragma unroll
      for (int i = 0; i < 4; ++i) {
        int gn = ntile + brr + 4 * i;
        int gk = kb + bkk;
        okB[i] = (gn < N) & (gk < K);
        long long o = okB[i] ? ((long long)gn * K + gk) : 0;
        vb[i] = Wt[o];
      }
#pragma unroll
      for (int i = 0; i < 16; ++i)
        As[(brr + 4 * i) * 40 + bkk] = (_Float16)(okA[i] ? va[i] : 0.f);
#pragma unroll
      for (int i = 0; i < 4; ++i)
        Bs[(brr + 4 * i) * 40 + bkk] = (_Float16)(okB[i] ? vb[i] : 0.f);
      __syncthreads();
      v16h afrag, bfrag;
#pragma unroll
      for (int e = 0; e < 16; ++e) {
        afrag[e] = As[(wave * 16 + l16) * 40 + a_k_of(e, kh)];
        bfrag[e] = Bs[l16 * 40 + a_k_of(e, kh)];
      }
      acc = __builtin_amdgcn_wmma_f32_16x16x32_f16(false, afrag, false, bfrag,
                                                   (short)0, acc, false, false);
      __syncthreads();
    }
  }
#pragma unroll
  for (int r = 0; r < 8; ++r) {
    int mm = mtile + kh * 8 + r;
    int nn = ntile + l16;
    if (mm < M && nn < N)
      out[(long long)mm * ldo + nn] = acc[r] + (bias ? bias[nn] : 0.f);
  }
}

// ---------------------------------------------------------------------------
// out = relu(o2 + [scale*skip+shift] or + identity)
// ---------------------------------------------------------------------------
__global__ void add_skip_relu_kernel(const float* __restrict__ o2, const float* __restrict__ sk,
                                     float* __restrict__ out,
                                     const float* __restrict__ scale,
                                     const float* __restrict__ shift,
                                     int total, int C, int HW2, int hasBN) {
  int i = blockIdx.x * blockDim.x + threadIdx.x;
  if (i >= total) return;
  float idn = sk[i];
  if (hasBN) {
    int c = (i / HW2) % C;
    idn = idn * scale[c] + shift[c];
  }
  float v = o2[i] + idn;
  out[i] = v > 0.f ? v : 0.f;
}

__global__ void maxpool_kernel(const float* __restrict__ in, float* __restrict__ out,
                               int Bn, int C, int H, int Wd) {
  int Ho = H >> 1, Wo = Wd >> 1;
  int total = Bn * C * Ho * Wo;
  int i = blockIdx.x * blockDim.x + threadIdx.x;
  if (i >= total) return;
  int x = i % Wo;
  int t = i / Wo;
  int y = t % Ho; t /= Ho;
  int c = t % C;
  int b = t / C;
  const float* p = in + ((long long)(b * C + c) * H + 2 * y) * Wd + 2 * x;
  out[i] = fmaxf(fmaxf(p[0], p[1]), fmaxf(p[Wd], p[Wd + 1]));
}

// mu[b,n] = dot(X[b,:], Wt[n,:]) + bias[n]   grid(Nout, B); float4 streaming
__global__ void fc_reduce_kernel(const float* __restrict__ X, const float* __restrict__ Wt,
                                 const float* __restrict__ bias, float* __restrict__ out,
                                 int Kin, int Nout) {
  __shared__ float s[256];
  int n = blockIdx.x, b = blockIdx.y, tid = threadIdx.x;
  const float4* xr = (const float4*)(X + (long long)b * Kin);
  const float4* wr = (const float4*)(Wt + (long long)n * Kin);
  int K4 = Kin >> 2;
  float a = 0.f;
  for (int k = tid; k < K4; k += 256) {
    float4 xv = xr[k], wv = wr[k];
    a += xv.x * wv.x + xv.y * wv.y + xv.z * wv.z + xv.w * wv.w;
  }
  s[tid] = a;
  __syncthreads();
  for (int st = 128; st > 0; st >>= 1) {
    if (tid < st) s[tid] += s[tid + st];
    __syncthreads();
  }
  if (tid == 0) out[b * Nout + n] = s[0] + (bias ? bias[n] : 0.f);
}

// out[b,j] = relu(dot(mu[b,:Z], W[j,:Z]) + bias[j]); K=64 via float4
__global__ void lin_relu_kernel(const float* __restrict__ mu, const float* __restrict__ W,
                                const float* __restrict__ bias, float* __restrict__ out,
                                int Z2, int Nout, int Bn) {
  int i = blockIdx.x * blockDim.x + threadIdx.x;
  int total = Bn * Nout;
  if (i >= total) return;
  int b = i / Nout;
  int j = i - b * Nout;
  const float4* wr = (const float4*)(W + (long long)j * Z2);
  const float4* mr = (const float4*)(mu + (long long)b * Z2);
  float a = bias[j];
  int K4 = Z2 >> 2;
  for (int k = 0; k < K4; ++k) {
    float4 wv = wr[k], mv = mr[k];
    a += mv.x * wv.x + mv.y * wv.y + mv.z * wv.z + mv.w * wv.w;
  }
  out[i] = a > 0.f ? a : 0.f;
}

// Ws[o,i] = sum_k W[k,o,i]
__global__ void sum_cheb_kernel(const float* __restrict__ W, float* __restrict__ Ws,
                                int Kc, int OI) {
  int i = blockIdx.x * blockDim.x + threadIdx.x;
  if (i >= OI) return;
  float a = 0.f;
  for (int k = 0; k < Kc; ++k) a += W[(long long)k * OI + i];
  Ws[i] = a;
}

// In-place relu(instance-norm over last dim). One wave32 per row.
__global__ void inorm_relu_kernel(float* __restrict__ h, int rows, int F) {
  int lane = threadIdx.x & 31;
  int row = blockIdx.x * (blockDim.x >> 5) + (threadIdx.x >> 5);
  if (row >= rows) return;
  float* p = h + (long long)row * F;
  float s = 0.f, s2 = 0.f;
  for (int f = lane; f < F; f += 32) {
    float v = p[f];
    s += v; s2 += v * v;
  }
#pragma unroll
  for (int off = 16; off > 0; off >>= 1) {
    s  += __shfl_xor(s, off, 32);
    s2 += __shfl_xor(s2, off, 32);
  }
  float m = s / (float)F;
  float var = s2 / (float)F - m * m;
  float inv = rsqrtf(var + EPSV);
  for (int f = lane; f < F; f += 32) {
    float v = (p[f] - m) * inv;
    p[f] = v > 0.f ? v : 0.f;
  }
}

// Small dense head: out[r,o] = dot(in[r,:K], Ws[o,:K]) + b
__global__ void cheb_small_kernel(const float* __restrict__ in, int lda,
                                  const float* __restrict__ Ws, const float* __restrict__ bias,
                                  float* __restrict__ out, int ldo, int rows, int K, int O) {
  int i = blockIdx.x * blockDim.x + threadIdx.x;
  int total = rows * O;
  if (i >= total) return;
  int r = i / O;
  int o = i - r * O;
  float a = bias ? bias[o] : 0.f;
  const float* p = in + (long long)r * lda;
  const float* wr = Ws + (long long)o * K;
  for (int k = 0; k < K; ++k) a += p[k] * wr[k];
  out[(long long)r * ldo + o] = a;
}

__global__ void copy_strided_kernel(const float* __restrict__ src, int lds_,
                                    float* __restrict__ dst, int ldd, int off,
                                    int rows, int F) {
  int i = blockIdx.x * blockDim.x + threadIdx.x;
  int total = rows * F;
  if (i >= total) return;
  int r = i / F;
  int f = i - r * F;
  dst[(long long)r * ldd + off + f] = src[(long long)r * lds_ + f];
}

// roi_align(aligned=True, 1x1 out, 3x3 window, 3x3 samples) -> dst[:,off:off+C]
__global__ void roi_kernel(const float* __restrict__ layer, const float* __restrict__ pos,
                           float* __restrict__ dst, int Bn, int N, int C, int H, int Wd,
                           int ldd, int off) {
  int i = blockIdx.x * blockDim.x + threadIdx.x;
  int total = Bn * N * C;
  if (i >= total) return;
  int c = i % C;
  int t = i / C;
  int n = t % N;
  int b = t / N;
  float px = pos[((long long)b * N + n) * 2 + 0] * (float)Wd;
  float py = pos[((long long)b * N + n) * 2 + 1] * (float)Wd;
  float sx = px - 1.5f;
  float sy = py - 1.5f;
  const float* img = layer + (long long)(b * C + c) * H * Wd;
  float sum = 0.f;
#pragma unroll
  for (int jy = 0; jy < 3; ++jy) {
    float yy = sy + (float)jy + 0.5f;
#pragma unroll
    for (int jx = 0; jx < 3; ++jx) {
      float xx = sx + (float)jx + 0.5f;
      if (yy >= -1.f && yy <= (float)H && xx >= -1.f && xx <= (float)Wd) {
        float y = fminf(fmaxf(yy, 0.f), (float)(H - 1));
        float x = fminf(fmaxf(xx, 0.f), (float)(Wd - 1));
        float y0 = floorf(y), x0 = floorf(x);
        int y0i = (int)y0, x0i = (int)x0;
        int y1i = min(y0i + 1, H - 1);
        int x1i = min(x0i + 1, Wd - 1);
        float ly = y - y0, lx = x - x0;
        float hy = 1.f - ly, hx = 1.f - lx;
        sum += hy * hx * img[y0i * Wd + x0i] + hy * lx * img[y0i * Wd + x1i] +
               ly * hx * img[y1i * Wd + x0i] + ly * lx * img[y1i * Wd + x1i];
      }
    }
  }
  dst[(long long)((long long)b * N + n) * ldd + off + c] = sum * (1.f / 9.f);
}

// out[b,r,f] = sum_{j<3} vals[3r+j] * h[b, cols[3r+j], f]
__global__ void pool_up_kernel(const float* __restrict__ h, const int* __restrict__ cols,
                               const float* __restrict__ vals, float* __restrict__ out,
                               int Bn, int N1c, int N2c, int F) {
  int i = blockIdx.x * blockDim.x + threadIdx.x;
  int total = Bn * N2c * F;
  if (i >= total) return;
  int f = i % F;
  int t = i / F;
  int r = t % N2c;
  int b = t / N2c;
  float a = 0.f;
#pragma unroll
  for (int j = 0; j < 3; ++j) {
    int e = 3 * r + j;
    a += vals[e] * h[((long long)b * N1c + cols[e]) * F + f];
  }
  out[i] = a;
}

// ---------------------------------------------------------------------------
// Host orchestration
// ---------------------------------------------------------------------------
extern "C" void kernel_launch(void* const* d_in, const int* in_sizes, int n_in,
                              void* d_out, int out_size, void* d_ws, size_t ws_size,
                              hipStream_t stream) {
  (void)in_sizes; (void)out_size; (void)ws_size;
  if (n_in < 84) return;
  const int Bq = 4;

  auto IN = [&](int i) { return (const float*)d_in[i]; };

  // ---- workspace carve (floats) ----
  float* ws = (float*)d_ws;
  size_t off = 0;
  auto alloc = [&](size_t n) {
    float* p = ws + off;
    off += (n + 63) & ~(size_t)63;
    return p;
  };
  float* bufA  = alloc(8388608);  // 4*8*512*512 (largest layer)
  float* bufB  = alloc(8388608);
  float* bufC  = alloc(8388608);
  float* H1    = alloc(2097152);
  float* H2    = alloc(2097152);
  float* conv5 = alloc(524288);   // [4,128,32,32]
  float* conv6 = alloc(131072);   // [4,128,16,16]
  float* muB   = alloc(256);
  float* pos1  = alloc(16384);    // [4,2048,2]
  float* pos2  = alloc(65536);    // [4,8192,2]
  float* WsB   = alloc(16384);    // largest summed cheb weight (128*128)
  float* sS1 = alloc(256); float* sT1 = alloc(256);
  float* sS2 = alloc(256); float* sT2 = alloc(256);
  float* sSS = alloc(256); float* sTS = alloc(256);

  // ---- helpers ----
  auto launch_stats = [&](const float* in_, const float* g_, const float* b_,
                          float* sc_, float* sh_, int C, int HW2) {
    bn_stats_fold_kernel<<<dim3(C), dim3(256), 0, stream>>>(in_, g_, b_, sc_, sh_, Bq, C, HW2);
  };
  auto launch_conv = [&](const float* in_, const float* w_, const float* bias_, float* out_,
                         const float* s_, const float* t_, int Cin, int H, int Wd, int Cout,
                         int ks, int pre) {
    int M = Bq * H * Wd;
    dim3 grid(ceilDiv(M, 16 * NWAVES), ceilDiv(Cout, 16));
    if (ks == 3)
      conv_wmma_kernel<3, 1><<<grid, dim3(32 * NWAVES), 0, stream>>>(
          in_, w_, bias_, out_, s_, t_, Bq, Cin, H, Wd, Cout, pre);
    else
      conv_wmma_kernel<1, 0><<<grid, dim3(32 * NWAVES), 0, stream>>>(
          in_, w_, bias_, out_, s_, t_, Bq, Cin, H, Wd, Cout, pre);
  };
  auto launch_pool = [&](const float* in_, float* out_, int C, int H, int Wd) {
    int total = Bq * C * (H / 2) * (Wd / 2);
    maxpool_kernel<<<dim3(ceilDiv(total, 256)), dim3(256), 0, stream>>>(in_, out_, Bq, C, H, Wd);
  };
  auto resblock = [&](const float* h, int Cin, int Cout, int H, int Wd, int pi, bool skip,
                      float* outBuf) {
    launch_stats(h, IN(pi + 0), IN(pi + 1), sS1, sT1, Cin, H * Wd);
    launch_conv(h, IN(pi + 2), IN(pi + 3), bufB, sS1, sT1, Cin, H, Wd, Cout, 3, 1);
    launch_stats(bufB, IN(pi + 4), IN(pi + 5), sS2, sT2, Cout, H * Wd);
    launch_conv(bufB, IN(pi + 6), IN(pi + 7), bufC, sS2, sT2, Cout, H, Wd, Cout, 3, 1);
    int total = Bq * Cout * H * Wd;
    if (skip) {
      launch_conv(h, IN(pi + 8), nullptr, bufB, nullptr, nullptr, Cin, H, Wd, Cout, 1, 0);
      launch_stats(bufB, IN(pi + 9), IN(pi + 10), sSS, sTS, Cout, H * Wd);
      add_skip_relu_kernel<<<dim3(ceilDiv(total, 256)), dim3(256), 0, stream>>>(
          bufC, bufB, outBuf, sSS, sTS, total, Cout, H * Wd, 1);
    } else {
      add_skip_relu_kernel<<<dim3(ceilDiv(total, 256)), dim3(256), 0, stream>>>(
          bufC, h, outBuf, nullptr, nullptr, total, Cout, H * Wd, 0);
    }
  };
  auto launch_sumW = [&](const float* W, int Kc, int OI) {
    sum_cheb_kernel<<<dim3(ceilDiv(OI, 256)), dim3(256), 0, stream>>>(W, WsB, Kc, OI);
  };
  auto launch_gemm = [&](const float* A, int lda, const float* Wt, const float* bias,
                         float* out_, int ldo, int M, int N, int K) {
    dim3 grid(ceilDiv(M, 16 * NWAVES), ceilDiv(N, 16));
    gemm_wmma_kernel<<<grid, dim3(32 * NWAVES), 0, stream>>>(A, lda, Wt, bias, out_, ldo, M, N, K);
  };
  auto launch_inorm = [&](float* h, int rows, int Ff) {
    inorm_relu_kernel<<<dim3(ceilDiv(rows, 8)), dim3(256), 0, stream>>>(h, rows, Ff);
  };
  auto launch_copy = [&](const float* src, int lds_, float* dst, int ldd, int o, int rows, int F) {
    copy_strided_kernel<<<dim3(ceilDiv(rows * F, 256)), dim3(256), 0, stream>>>(
        src, lds_, dst, ldd, o, rows, F);
  };

  // ---- encoder ----
  const float* x = IN(0);
  resblock(x, 1, 8, 512, 512, 1, true, bufA);
  launch_pool(bufA, H1, 8, 512, 512);
  resblock(H1, 8, 16, 256, 256, 12, true, bufA);
  launch_pool(bufA, H2, 16, 256, 256);
  resblock(H2, 16, 32, 128, 128, 23, true, bufA);
  launch_pool(bufA, H1, 32, 128, 128);
  resblock(H1, 32, 64, 64, 64, 34, true, bufA);
  launch_pool(bufA, H2, 64, 64, 64);
  resblock(H2, 64, 128, 32, 32, 45, true, conv5);
  launch_pool(conv5, H1, 128, 32, 32);
  resblock(H1, 128, 128, 16, 16, 56, false, conv6);

  // ---- latent ----
  fc_reduce_kernel<<<dim3(64, Bq), dim3(256), 0, stream>>>(conv6, IN(64), IN(65), muB, 32768, 64);
  {
    int total = Bq * 262144;
    lin_relu_kernel<<<dim3(ceilDiv(total, 256)), dim3(256), 0, stream>>>(
        muB, IN(66), IN(67), bufA, 64, 262144, Bq);
  }

  // ---- decoder (N1 = 2048 nodes, then N2 = 8192) ----
  const int R1 = Bq * 2048;   // 8192 rows
  const int R2 = Bq * 8192;   // 32768 rows

  // g6: 128 -> 128
  launch_sumW(IN(68), 6, 128 * 128);
  launch_gemm(bufA, 128, WsB, IN(69), bufB, 128, R1, 128, 128);
  launch_inorm(bufB, R1, 128);
  // g5: 128 -> 64
  launch_sumW(IN(70), 6, 64 * 128);
  launch_gemm(bufB, 128, WsB, IN(71), bufA, 64, R1, 64, 128);
  launch_inorm(bufA, R1, 64);
  // pos1 = cheb(h, sc1_W)  (cheb order 1 -> weights used directly)
  cheb_small_kernel<<<dim3(ceilDiv(R1 * 2, 256)), dim3(256), 0, stream>>>(
      bufA, 64, IN(72), nullptr, pos1, 2, R1, 64, 2);
  // concat [h(64) | roi(conv6,128) | pos1(2)] -> bufC stride 194
  launch_copy(bufA, 64, bufC, 194, 0, R1, 64);
  roi_kernel<<<dim3(ceilDiv(Bq * 2048 * 128, 256)), dim3(256), 0, stream>>>(
      conv6, pos1, bufC, Bq, 2048, 128, 16, 16, 194, 64);
  launch_copy(pos1, 2, bufC, 194, 192, R1, 2);
  // g4: 194 -> 64
  launch_sumW(IN(73), 6, 64 * 194);
  launch_gemm(bufC, 194, WsB, IN(74), bufA, 64, R1, 64, 194);
  launch_inorm(bufA, R1, 64);
  // pool_up: N1 -> N2 (deterministic fanout-3 gather)
  pool_up_kernel<<<dim3(ceilDiv(Bq * 8192 * 64, 256)), dim3(256), 0, stream>>>(
      bufA, (const int*)d_in[83], (const float*)d_in[81], bufB, Bq, 2048, 8192, 64);
  // g3: 64 -> 32
  launch_sumW(IN(75), 6, 32 * 64);
  launch_gemm(bufB, 64, WsB, IN(76), bufA, 32, R2, 32, 64);
  launch_inorm(bufA, R2, 32);
  // pos2
  cheb_small_kernel<<<dim3(ceilDiv(R2 * 2, 256)), dim3(256), 0, stream>>>(
      bufA, 32, IN(77), nullptr, pos2, 2, R2, 32, 2);
  // concat [h(32) | roi(conv5,128) | pos2(2)] -> bufC stride 162
  launch_copy(bufA, 32, bufC, 162, 0, R2, 32);
  roi_kernel<<<dim3(ceilDiv(Bq * 8192 * 128, 256)), dim3(256), 0, stream>>>(
      conv5, pos2, bufC, Bq, 8192, 128, 32, 32, 162, 32);
  launch_copy(pos2, 2, bufC, 162, 160, R2, 2);
  // g2: 162 -> 16
  launch_sumW(IN(78), 6, 16 * 162);
  launch_gemm(bufC, 162, WsB, IN(79), bufB, 16, R2, 16, 162);
  launch_inorm(bufB, R2, 16);
  // g1: 16 -> 2 (no bias) -> output [4, 8192, 2]
  cheb_small_kernel<<<dim3(ceilDiv(R2 * 2, 256)), dim3(256), 0, stream>>>(
      bufB, 16, IN(80), nullptr, (float*)d_out, 2, R2, 16, 2);
}
